// SimpleMultiLabelConstrativeLoss_33268816675395
// MI455X (gfx1250) — matrix operand order repaired and checked by reading, not compiled, not run on hardware
//
#include <hip/hip_runtime.h>
#include <hip/hip_bf16.h>

typedef __attribute__((ext_vector_type(16))) _Float16 v16h;
typedef __attribute__((ext_vector_type(8)))  _Float16 v8h;
typedef __attribute__((ext_vector_type(8)))  float    v8f;

#define CLS_PAD     16
#define NUM_CLASSES 10
#define POS_MARGIN  0.4f
#define NEG_MARGIN  1.0f
#define LEPS        1e-8f

// ---------------------------------------------------------------------------
// Kernel 0: zero the accumulators in workspace (S32, counts, loss)
// ---------------------------------------------------------------------------
__global__ void init_ws_kernel(float* __restrict__ S32, float* __restrict__ counts,
                               float* __restrict__ lossAcc, int sTotal) {
    int i = blockIdx.x * blockDim.x + threadIdx.x;
    for (int idx = i; idx < sTotal; idx += blockDim.x * gridDim.x) S32[idx] = 0.f;
    if (i < CLS_PAD) counts[i] = 0.f;
    if (i == 0) lossAcc[0] = 0.f;
}

// ---------------------------------------------------------------------------
// Kernel 1: row-normalize x -> xn16 (f16), accumulate per-class sums + counts.
// One wave32 per row, 8 waves per block, 64 rows per block. LDS class-sum
// staging (ds_add_f32) then one global-atomic flush per block.
// Assumes D == 512.
// ---------------------------------------------------------------------------
__global__ void norm_accum_kernel(const float* __restrict__ x,
                                  const int*   __restrict__ labels,
                                  _Float16*    __restrict__ xn16,
                                  float*       __restrict__ S32,
                                  float*       __restrict__ counts,
                                  int N, int D) {
    __shared__ float sS[CLS_PAD * 512];
    __shared__ float sCnt[CLS_PAD];
    const int tid  = threadIdx.x;
    const int lane = tid & 31;
    const int wave = tid >> 5;

    for (int i = tid; i < CLS_PAD * D; i += blockDim.x) sS[i] = 0.f;
    if (tid < CLS_PAD) sCnt[tid] = 0.f;
    __syncthreads();

    const int rowsPerBlock = 64;
    const int rowsPerWave  = rowsPerBlock / 8;
    const int rowBase = blockIdx.x * rowsPerBlock + wave * rowsPerWave;

    for (int r = 0; r < rowsPerWave; ++r) {
        const int row = rowBase + r;
        const float* xr = x + (size_t)row * D;
        float val[16];
        float ss = 0.f;
        #pragma unroll
        for (int j = 0; j < 16; ++j) {          // D = 512 = 16 * 32 lanes
            val[j] = xr[j * 32 + lane];
            ss += val[j] * val[j];
        }
        #pragma unroll
        for (int m = 16; m >= 1; m >>= 1) ss += __shfl_xor(ss, m, 32);
        const float inv = 1.0f / fmaxf(sqrtf(ss), LEPS);   // eps-clamped norm
        const int lbl = labels[row];
        _Float16* xo = xn16 + (size_t)row * D;
        #pragma unroll
        for (int j = 0; j < 16; ++j) {
            const float xv  = val[j] * inv;
            const int   col = j * 32 + lane;
            xo[col] = (_Float16)xv;
            atomicAdd(&sS[lbl * D + col], xv);             // ds_add_f32
        }
        if (lane == 0) atomicAdd(&sCnt[lbl], 1.0f);
    }
    __syncthreads();
    for (int i = tid; i < CLS_PAD * D; i += blockDim.x) {
        float v = sS[i];
        if (v != 0.f) atomicAdd(&S32[i], v);
    }
    if (tid < CLS_PAD) {
        float v = sCnt[tid];
        if (v != 0.f) atomicAdd(&counts[tid], v);
    }
}

// ---------------------------------------------------------------------------
// Kernel 2: convert class sums to f16 for the WMMA B-operand (rows 10..15 = 0)
// ---------------------------------------------------------------------------
__global__ void cvt_s_kernel(const float* __restrict__ S32,
                             _Float16* __restrict__ S16, int total) {
    int i = blockIdx.x * blockDim.x + threadIdx.x;
    if (i < total) S16[i] = (_Float16)S32[i];
}

// ---------------------------------------------------------------------------
// Kernel 3: G = xn * S^T via v_wmma_f32_16x16x32_f16, fused loss epilogue.
// One wave per 16-row output tile; 8 waves/block -> 128 rows per block.
// ---------------------------------------------------------------------------
__global__ void wmma_loss_kernel(const _Float16* __restrict__ xn16,
                                 const _Float16* __restrict__ S16,
                                 const float*    __restrict__ counts,
                                 const int*      __restrict__ labels,
                                 float*          __restrict__ lossAcc,
                                 int N, int D) {
    const int tid  = threadIdx.x;
    const int lane = tid & 31;
    const int wave = tid >> 5;
    const int tile = blockIdx.x * 8 + wave;
    const int rowBase = tile * 16;

    const int hi   = (lane >= 16) ? 1 : 0;
    const int aRow = rowBase + (lane & 15);  // A: lane holds row M = lane%16
    const int aOff = hi * 8;                 // A halves 0..7 -> K = aOff+0..7, 8..15 -> K = 16+aOff+0..7
    const int bOff = hi * 8;                 // B: lane holds col N = lane%16, halves 0..7 -> K = bOff+0..7, 8..15 -> K = 16+bOff+0..7
    const int n    = lane & 15;              // class column

    v8f acc = {};
    for (int k0 = 0; k0 < D; k0 += 32) {
        const _Float16* pa = xn16 + (size_t)aRow * D + k0 + aOff;
        v8h a0 = *(const v8h*)(pa);
        v8h a1 = *(const v8h*)(pa + 16);
        const _Float16* pb = S16 + (size_t)n * D + k0 + bOff;
        v8h b0 = *(const v8h*)(pb);
        v8h b1 = *(const v8h*)(pb + 16);
        v16h a = __builtin_shufflevector(a0, a1, 0,1,2,3,4,5,6,7,8,9,10,11,12,13,14,15);
        v16h b = __builtin_shufflevector(b0, b1, 0,1,2,3,4,5,6,7,8,9,10,11,12,13,14,15);
        acc = __builtin_amdgcn_wmma_f32_16x16x32_f16(
            /*neg_a=*/false, a, /*neg_b=*/false, b,
            /*c_mod=*/(short)0, acc, /*reuse_a=*/false, /*reuse_b=*/false);
    }

    // Epilogue: D-matrix VGPR v holds (M = v + 8*hi, N = lane%16)
    float partial = 0.f;
    if (n < NUM_CLASSES) {
        const float cnt    = counts[n];
        const float invDen = 1.0f / (cnt + LEPS);
        const int   mBase  = hi * 8;
        #pragma unroll
        for (int v = 0; v < 8; ++v) {
            const int   row = rowBase + mBase + v;
            const int   lbl = labels[row];
            const float da  = (cnt - acc[v]) * invDen;   // dist_avg[row][n]
            if (lbl == n) {
                float t = fmaxf(POS_MARGIN - da, 0.f);
                partial += t * t + 1.0f;                           // + relu(1-0)^2
            } else {
                float t = fmaxf(NEG_MARGIN - da, 0.f);
                partial += t * t + POS_MARGIN * POS_MARGIN;        // + relu(0.4-0)^2
            }
        }
    }
    #pragma unroll
    for (int m = 16; m >= 1; m >>= 1) partial += __shfl_xor(partial, m, 32);
    if (lane == 0) atomicAdd(lossAcc, partial);
}

// ---------------------------------------------------------------------------
// Kernel 4: finalize: out = loss_sum / N
// ---------------------------------------------------------------------------
__global__ void finalize_kernel(const float* __restrict__ lossAcc,
                                float* __restrict__ out, float invN) {
    if (blockIdx.x == 0 && threadIdx.x == 0) out[0] = lossAcc[0] * invN;
}

// ---------------------------------------------------------------------------
extern "C" void kernel_launch(void* const* d_in, const int* in_sizes, int n_in,
                              void* d_out, int out_size, void* d_ws, size_t ws_size,
                              hipStream_t stream) {
    const float* x      = (const float*)d_in[0];   // [N, D] f32
    const int*   labels = (const int*)d_in[1];     // [N] i32
    float*       out    = (float*)d_out;           // scalar f32

    const int N = in_sizes[1];              // 8192
    const int D = in_sizes[0] / N;          // 512

    // Workspace layout
    char* ws = (char*)d_ws;
    const size_t xnBytes  = (size_t)N * D * sizeof(_Float16);        // 8 MiB
    const size_t s32Bytes = (size_t)CLS_PAD * D * sizeof(float);     // 32 KiB
    const size_t s16Bytes = (size_t)CLS_PAD * D * sizeof(_Float16);  // 16 KiB
    _Float16* xn16   = (_Float16*)ws;
    float*    S32    = (float*)(ws + xnBytes);
    _Float16* S16    = (_Float16*)(ws + xnBytes + s32Bytes);
    float*    counts = (float*)(ws + xnBytes + s32Bytes + s16Bytes);
    float*    lossAcc = counts + CLS_PAD;

    init_ws_kernel<<<8, 256, 0, stream>>>(S32, counts, lossAcc, CLS_PAD * D);
    norm_accum_kernel<<<N / 64, 256, 0, stream>>>(x, labels, xn16, S32, counts, N, D);
    cvt_s_kernel<<<(CLS_PAD * D + 255) / 256, 256, 0, stream>>>(S32, S16, CLS_PAD * D);
    wmma_loss_kernel<<<N / 128, 256, 0, stream>>>(xn16, S16, counts, labels, lossAcc, N, D);
    finalize_kernel<<<1, 64, 0, stream>>>(lossAcc, out, 1.0f / (float)N);
}
